// DenseBiDecoder_1906965479573
// MI455X (gfx1250) — matrix operand order repaired and checked by reading, not compiled, not run on hardware
//
#include <hip/hip_runtime.h>

// ---------------------------------------------------------------------------
// DenseBiDecoder for MI455X (gfx1250): logits[e,c] = sum_b W[c,b] * u_e^T P_b i_e
// bf16 WMMA (v_wmma_f32_16x16x32_bf16) for the u @ P_b GEMM, P resident in LDS
// (256KB bf16; CDNA5's 320KB/WGP makes the whole operand fit), f32 everywhere
// else. One wave processes one 16-edge tile at a time.
//
// Round-2 fix: keep the nt loop ROLLED (#pragma unroll 1). Full unrolling made
// the scheduler hoist all 512 ds_loads and spill B operands to scratch.
// ---------------------------------------------------------------------------

typedef __bf16 bf16_t;                                     // storage only
typedef bf16_t v16bf __attribute__((ext_vector_type(16))); // WMMA A/B operand
typedef float  v8f   __attribute__((ext_vector_type(8)));  // WMMA C/D operand

#define D_DIM   256
#define NBASIS  2
#define NCLS    8
#define PT_DWORDS (NBASIS * D_DIM * D_DIM / 2)   // 65536 dwords = 256 KB bf16

__device__ __forceinline__ unsigned pack_bf16x2(float lo, float hi) {
  // round-to-nearest-even f32 -> bf16, packed two per dword (lo in [15:0])
  unsigned a = __float_as_uint(lo);
  unsigned b = __float_as_uint(hi);
  a += 0x7fffu + ((a >> 16) & 1u);
  b += 0x7fffu + ((b >> 16) & 1u);
  return (b & 0xffff0000u) | (a >> 16);
}

// -------- Kernel 1: P[b][k][n] f32  ->  Pt[b][n][k] bf16 (K-major rows) -----
// Transposed so each WMMA lane's B column is 16 K-contiguous bf16 = 32 bytes,
// in the same k = 16*hf + s order as the A gather (2 x ds_load_b128).
__global__ void prep_Pt_kernel(const float* __restrict__ P,
                               unsigned* __restrict__ Pt) {
  const int t = blockIdx.x * blockDim.x + threadIdx.x;
  if (t >= PT_DWORDS) return;
  const int n  = t & (D_DIM - 1);          // coalesced reads over n
  const int kp = (t >> 8) & (D_DIM / 2 - 1);
  const int b  = t >> 15;
  const int k0 = kp << 1;
  const float f0 = P[((b * D_DIM) + k0    ) * D_DIM + n];
  const float f1 = P[((b * D_DIM) + k0 + 1) * D_DIM + n];
  Pt[(b * D_DIM + n) * (D_DIM / 2) + kp] = pack_bf16x2(f0, f1);
}

// -------- Kernel 2: fused gather + bilinear WMMA + combine ------------------
__global__ void __launch_bounds__(256, 1)
bidecoder_kernel(const float* __restrict__ h_src,
                 const float* __restrict__ h_dst,
                 const int*   __restrict__ u_idx,
                 const int*   __restrict__ v_idx,
                 const unsigned* __restrict__ Pt,   // bf16 [NB][n][D] packed
                 const float* __restrict__ W,       // [NC][NB]
                 float* __restrict__ out,           // [E][NC]
                 int n_edges) {
  extern __shared__ unsigned char smem[];
  unsigned* ldsPt = (unsigned*)smem;                // 65536 dwords = 256 KB

  // Stage the full (bf16) P operand into LDS once per workgroup.
  {
    const uint4* src = (const uint4*)Pt;
    uint4*       dst = (uint4*)ldsPt;
    const int nvec = PT_DWORDS / 4;                 // 16384 x 16B
    for (int i = threadIdx.x; i < nvec; i += blockDim.x) dst[i] = src[i];
  }
  __syncthreads();

  const int lane = threadIdx.x & 31;
  const int nl   = lane & 15;       // A-row m / B,N column / C column
  const int hf   = lane >> 4;       // half-wave selector for K/M splits

  // NB->NC combine weights (tiny, uniform -> SGPRs)
  float w[NCLS * NBASIS];
#pragma unroll
  for (int i = 0; i < NCLS * NBASIS; ++i) w[i] = W[i];

  const int wave    = blockIdx.x * (blockDim.x >> 5) + (threadIdx.x >> 5);
  const int nwave   = gridDim.x * (blockDim.x >> 5);
  const int ntile_e = n_edges >> 4;                 // 16 edges per tile

  for (int t = wave; t < ntile_e; t += nwave) {
    const int e0 = t << 4;

    // Prefetch next tile's index cachelines (global_prefetch_b8).
    const int tn = t + nwave;
    if (tn < ntile_e) {
      __builtin_prefetch(u_idx + (tn << 4), 0, 0);
      __builtin_prefetch(v_idx + (tn << 4), 0, 0);
    }

    // ---- A: gather ufeat rows straight into WMMA 16-bit A layout ----------
    // lane l: row m = l&15; V0-7 hold K = hf*16+{0..15}
    const int    mu   = u_idx[e0 + nl];
    const float* arow = h_src + (long)mu * D_DIM;
    v16bf A[8];
#pragma unroll
    for (int kc = 0; kc < 8; ++kc) {
      const float* p0 = arow + kc * 32 + hf * 16;
      const float4 f0 = *(const float4*)(p0);
      const float4 f1 = *(const float4*)(p0 + 4);
      const float4 f2 = *(const float4*)(p0 + 8);
      const float4 f3 = *(const float4*)(p0 + 12);
      union { v16bf v; unsigned d[8]; } a;
      a.d[0] = pack_bf16x2(f0.x, f0.y);
      a.d[1] = pack_bf16x2(f0.z, f0.w);
      a.d[2] = pack_bf16x2(f1.x, f1.y);
      a.d[3] = pack_bf16x2(f1.z, f1.w);
      a.d[4] = pack_bf16x2(f2.x, f2.y);
      a.d[5] = pack_bf16x2(f2.z, f2.w);
      a.d[6] = pack_bf16x2(f3.x, f3.y);
      a.d[7] = pack_bf16x2(f3.z, f3.w);
      A[kc] = a.v;
    }

    // ---- ifeat row base pointers for the C/D layout rows m = v + 8*hf -----
    const float* irow[8];
    {
      const int4 i0 = *(const int4*)(v_idx + e0 + hf * 8);
      const int4 i1 = *(const int4*)(v_idx + e0 + hf * 8 + 4);
      irow[0] = h_dst + (long)i0.x * D_DIM;
      irow[1] = h_dst + (long)i0.y * D_DIM;
      irow[2] = h_dst + (long)i0.z * D_DIM;
      irow[3] = h_dst + (long)i0.w * D_DIM;
      irow[4] = h_dst + (long)i1.x * D_DIM;
      irow[5] = h_dst + (long)i1.y * D_DIM;
      irow[6] = h_dst + (long)i1.z * D_DIM;
      irow[7] = h_dst + (long)i1.w * D_DIM;
    }

    float acc[NBASIS][8];
#pragma unroll
    for (int b = 0; b < NBASIS; ++b)
#pragma unroll
      for (int v = 0; v < 8; ++v) acc[b][v] = 0.0f;

    // ---- N-tile loop: C_b = A @ P_b[:, nt-tile], fuse C_b .* ifeat --------
    // KEPT ROLLED: one iteration = 16 WMMAs + 32 ds_load_b128 + 8 b32 gathers;
    // full unroll previously forced B-operand spills to scratch.
#pragma unroll 1
    for (int nt = 0; nt < 16; ++nt) {
      const int ncol = (nt << 4) + nl;
      float iv[8];
#pragma unroll
      for (int v = 0; v < 8; ++v) iv[v] = irow[v][ncol];   // f32, C-layout

      // lane's B column dword offsets: row (b*256 + nt*16 + nl), K-half = hf
      const unsigned r0off =
          ((unsigned)((nt << 4) + nl) << 7) + (unsigned)(hf << 3);
      const unsigned r1off = r0off + (unsigned)(D_DIM << 7);  // basis 1

      v8f c0 = {0.f, 0.f, 0.f, 0.f, 0.f, 0.f, 0.f, 0.f};
      v8f c1 = {0.f, 0.f, 0.f, 0.f, 0.f, 0.f, 0.f, 0.f};
#pragma unroll
      for (int kc = 0; kc < 8; ++kc) {
        // interleave bases: each B is consumed right after its 2x ds_load_b128,
        // and the two independent accumulation chains hide WMMA RAW latency.
        const v16bf b0 = *(const v16bf*)(ldsPt + r0off + (kc << 4));
        c0 = __builtin_amdgcn_wmma_f32_16x16x32_bf16(
            false, A[kc], false, b0, (short)0, c0, false, false);
        const v16bf b1 = *(const v16bf*)(ldsPt + r1off + (kc << 4));
        c1 = __builtin_amdgcn_wmma_f32_16x16x32_bf16(
            false, A[kc], false, b1, (short)0, c1, false, false);
      }
#pragma unroll
      for (int v = 0; v < 8; ++v) {
        acc[0][v] += c0[v] * iv[v];
        acc[1][v] += c1[v] * iv[v];
      }
    }

    // ---- reduce the 16 N-columns living across the 16 lanes of each half --
#pragma unroll
    for (int v = 0; v < 8; ++v) {
      float r0 = acc[0][v], r1 = acc[1][v];
#pragma unroll
      for (int m = 1; m <= 8; m <<= 1) {
        r0 += __shfl_xor(r0, m, 32);
        r1 += __shfl_xor(r1, m, 32);
      }
      acc[0][v] = r0;
      acc[1][v] = r1;
    }

    // ---- combine NB->NC and store logits (edge m = v + 8*hf) --------------
#pragma unroll
    for (int v = 0; v < 8; ++v) {
      if (nl == v) {
        const long  e  = (long)e0 + (hf << 3) + v;
        const float o0 = acc[0][v];
        const float o1 = acc[1][v];
        float4 r0, r1;
        r0.x = o0 * w[0]  + o1 * w[1];
        r0.y = o0 * w[2]  + o1 * w[3];
        r0.z = o0 * w[4]  + o1 * w[5];
        r0.w = o0 * w[6]  + o1 * w[7];
        r1.x = o0 * w[8]  + o1 * w[9];
        r1.y = o0 * w[10] + o1 * w[11];
        r1.z = o0 * w[12] + o1 * w[13];
        r1.w = o0 * w[14] + o1 * w[15];
        float4* op = (float4*)(out + e * NCLS);
        op[0] = r0;
        op[1] = r1;
      }
    }
  }
}

// ---------------------------------------------------------------------------
extern "C" void kernel_launch(void* const* d_in, const int* in_sizes, int n_in,
                              void* d_out, int out_size, void* d_ws, size_t ws_size,
                              hipStream_t stream) {
  const float* h_src = (const float*)d_in[0];
  const float* h_dst = (const float*)d_in[1];
  const int*   u_idx = (const int*)d_in[2];
  const int*   v_idx = (const int*)d_in[3];
  const float* P     = (const float*)d_in[4];
  const float* W     = (const float*)d_in[5];
  float*       out   = (float*)d_out;
  unsigned*    Pt    = (unsigned*)d_ws;        // 256 KB bf16 transposed P
  const int n_edges  = in_sizes[2];

  // 1) convert + transpose P to bf16 once (L2-resident thereafter)
  prep_Pt_kernel<<<PT_DWORDS / 256, 256, 0, stream>>>(P, Pt);

  // 2) fused gather + bilinear WMMA + combine; 256 KB dynamic LDS per WG
  //    (1 WG/WGP, 8 wave32 per WG; each wave owns independent 16-edge tiles)
  bidecoder_kernel<<<256, 256, PT_DWORDS * 4, stream>>>(
      h_src, h_dst, u_idx, v_idx, Pt, W, out, n_edges);
}
